// AttentionSubsample_51470888075476
// MI455X (gfx1250) — matrix operand order repaired and checked by reading, not compiled, not run on hardware
//
#include <hip/hip_runtime.h>
#include <hip/hip_bf16.h>
#include <cstddef>

// ---------------- types ----------------
typedef _Float16 v16h __attribute__((ext_vector_type(16)));
typedef _Float16 v8h  __attribute__((ext_vector_type(8)));
typedef float    v8f  __attribute__((ext_vector_type(8)));

// ---------------- problem constants ----------------
#define BATCH   64
#define NTOK    1280            // tokens per batch (32*32 + 16*16)
#define NQTOK   320             // subsampled query tokens (16*16 + 8*8)
#define INDIM   256
#define HEADS   8
#define KD      32              // key dim per head
#define DV      64              // value dim per head
#define CKV     768             // H*(KD+DV)
#define CQ      256             // H*KD
#define COUT    512
#define DH      512             // H*DV
#define NT      (BATCH*NTOK)    // 81920 rows for KV gemm
#define NQ      (BATCH*NQTOK)   // 20480 rows for Q / out gemms
#define SCALE   0.17677669529663687f   // 32^-0.5
#define EPS     1e-5f

// ---------------- WMMA wrapper ----------------
__device__ __forceinline__ v8f wmma_f16(v16h a, v16h b, v8f c) {
    return __builtin_amdgcn_wmma_f32_16x16x32_f16(false, a, false, b, (short)0, c,
                                                  false, false);
}

// ---------------- fragment loaders ----------------
// A 16x32 f16 fragment from a row-major f32 row pointer (points at row's K-chunk)
__device__ __forceinline__ v16h load_a_f32(const float* p, int hi) {
    v16h a;
    const float4* q0 = (const float4*)(p + 8 * hi);
    float4 f0 = q0[0], f1 = q0[1];
    const float4* q1 = (const float4*)(p + 16 + 8 * hi);
    float4 f2 = q1[0], f3 = q1[1];
    a[0] = (_Float16)f0.x; a[1] = (_Float16)f0.y; a[2] = (_Float16)f0.z; a[3] = (_Float16)f0.w;
    a[4] = (_Float16)f1.x; a[5] = (_Float16)f1.y; a[6] = (_Float16)f1.z; a[7] = (_Float16)f1.w;
    a[8]  = (_Float16)f2.x; a[9]  = (_Float16)f2.y; a[10] = (_Float16)f2.z; a[11] = (_Float16)f2.w;
    a[12] = (_Float16)f3.x; a[13] = (_Float16)f3.y; a[14] = (_Float16)f3.z; a[15] = (_Float16)f3.w;
    return a;
}
// B 32x16 f16 fragment from a row-major f32 row pointer: K = 8*hi+0..7, 16+8*hi+0..7
__device__ __forceinline__ v16h load_b_f32(const float* p, int hi) {
    v16h b;
    const float4* q = (const float4*)(p + 8 * hi);
    float4 f0 = q[0], f1 = q[1];
    const float4* q2 = (const float4*)(p + 16 + 8 * hi);
    float4 f2 = q2[0], f3 = q2[1];
    b[0] = (_Float16)f0.x; b[1] = (_Float16)f0.y; b[2] = (_Float16)f0.z; b[3] = (_Float16)f0.w;
    b[4] = (_Float16)f1.x; b[5] = (_Float16)f1.y; b[6] = (_Float16)f1.z; b[7] = (_Float16)f1.w;
    b[8]  = (_Float16)f2.x; b[9]  = (_Float16)f2.y; b[10] = (_Float16)f2.z; b[11] = (_Float16)f2.w;
    b[12] = (_Float16)f3.x; b[13] = (_Float16)f3.y; b[14] = (_Float16)f3.z; b[15] = (_Float16)f3.w;
    return b;
}
// A 16x32 f16 fragment directly from f16 memory
__device__ __forceinline__ v16h load_a_f16(const _Float16* p, int hi) {
    v16h a;
    v8h lo = *(const v8h*)(p + 8 * hi);
    v8h hi8 = *(const v8h*)(p + 16 + 8 * hi);
#pragma unroll
    for (int i = 0; i < 8; ++i) { a[i] = lo[i]; a[8 + i] = hi8[i]; }
    return a;
}

// ---------------- small utility kernels ----------------
__global__ void zero_f32(float* p, int n) {
    int i = blockIdx.x * blockDim.x + threadIdx.x;
    if (i < n) p[i] = 0.f;
}

// fold BN stats into per-channel scale/bias
__global__ void bn_finalize(const float* __restrict__ sum, const float* __restrict__ sq,
                            const float* __restrict__ gamma, const float* __restrict__ beta,
                            float* __restrict__ scale, float* __restrict__ bias,
                            int n, float inv_cnt) {
    int i = blockIdx.x * blockDim.x + threadIdx.x;
    if (i >= n) return;
    float m = sum[i] * inv_cnt;
    float v = sq[i] * inv_cnt - m * m;
    float rs = rsqrtf(v + EPS);
    float s = gamma[i] * rs;
    scale[i] = s;
    bias[i] = beta[i] - m * s;
}

__global__ void bn_apply(float* __restrict__ out, const float* __restrict__ scale,
                         const float* __restrict__ bias) {
    int i = blockIdx.x * blockDim.x + threadIdx.x;   // NQ*COUT total
    int c = i & (COUT - 1);
    out[i] = out[i] * scale[c] + bias[c];
}

// ---------------- stats helper (per-block LDS reduction -> global atomics) ----
__device__ __forceinline__ void accum_stats(const v8f* acc, int lane, int n0,
                                            float* s_sum, float* s_sq,
                                            float* g_sum, float* g_sq) {
    if (threadIdx.x < 64) { s_sum[threadIdx.x] = 0.f; s_sq[threadIdx.x] = 0.f; }
    __syncthreads();
#pragma unroll
    for (int j = 0; j < 4; ++j) {
        float ps = 0.f, pq = 0.f;
#pragma unroll
        for (int r = 0; r < 8; ++r) { float v = acc[j][r]; ps += v; pq += v * v; }
        int ch = 16 * j + (lane & 15);
        atomicAdd(&s_sum[ch], ps);
        atomicAdd(&s_sq[ch], pq);
    }
    __syncthreads();
    if (threadIdx.x < 64) {
        atomicAdd(&g_sum[n0 + threadIdx.x], s_sum[threadIdx.x]);
        atomicAdd(&g_sq[n0 + threadIdx.x], s_sq[threadIdx.x]);
    }
}

// ---------------- GEMM 1: KV = X[NT,256] * W_kv^T -> f16 [NT,768] + stats -----
__global__ __launch_bounds__(256)
void gemm_kv(const float* __restrict__ X, const float* __restrict__ W,
             _Float16* __restrict__ out, float* __restrict__ g_sum, float* __restrict__ g_sq) {
    __shared__ float s_sum[64], s_sq[64];
    int wave = threadIdx.x >> 5, lane = threadIdx.x & 31, hi = lane >> 4;
    int m0 = blockIdx.x * 128 + wave * 16;
    int n0 = blockIdx.y * 64;
    int arow = m0 + (lane & 15);
    const float* aptr = X + (size_t)arow * INDIM;
    v8f acc[4] = {};
#pragma unroll 2
    for (int kk = 0; kk < INDIM; kk += 32) {
        // prefetch next A chunk of the big X stream into cache
        if (kk + 32 < INDIM) __builtin_prefetch(aptr + kk + 32, 0, 1);
        // materialize A + all four B fragments in distinct registers first:
        // the four WMMAs below then have no mutual register hazards.
        v16h a = load_a_f32(aptr + kk, hi);
        v16h b0 = load_b_f32(W + (size_t)(n0 +  0 + (lane & 15)) * INDIM + kk, hi);
        v16h b1 = load_b_f32(W + (size_t)(n0 + 16 + (lane & 15)) * INDIM + kk, hi);
        v16h b2 = load_b_f32(W + (size_t)(n0 + 32 + (lane & 15)) * INDIM + kk, hi);
        v16h b3 = load_b_f32(W + (size_t)(n0 + 48 + (lane & 15)) * INDIM + kk, hi);
        acc[0] = wmma_f16(a, b0, acc[0]);
        acc[1] = wmma_f16(a, b1, acc[1]);
        acc[2] = wmma_f16(a, b2, acc[2]);
        acc[3] = wmma_f16(a, b3, acc[3]);
    }
    accum_stats(acc, lane, n0, s_sum, s_sq, g_sum, g_sq);
#pragma unroll
    for (int j = 0; j < 4; ++j) {
        int n = n0 + 16 * j + (lane & 15);
#pragma unroll
        for (int r = 0; r < 8; ++r) {
            int m = m0 + r + 8 * hi;
            out[(size_t)m * CKV + n] = (_Float16)acc[j][r];
        }
    }
}

// ---------------- GEMM 2: Q = subsample(X)[NQ,256] * W_q^T -> f16 [NQ,256] ----
__device__ __forceinline__ int sub_token(int qi) {
    if (qi < 256) { int qr = qi >> 4, qc = qi & 15; return 64 * qr + 2 * qc; }
    int t = qi - 256; int zr = t >> 3, zc = t & 7;
    return 1024 + 32 * zr + 2 * zc;
}

__global__ __launch_bounds__(256)
void gemm_q(const float* __restrict__ X, const float* __restrict__ W,
            _Float16* __restrict__ out, float* __restrict__ g_sum, float* __restrict__ g_sq) {
    __shared__ float s_sum[64], s_sq[64];
    int wave = threadIdx.x >> 5, lane = threadIdx.x & 31, hi = lane >> 4;
    int m0 = blockIdx.x * 128 + wave * 16;
    int n0 = blockIdx.y * 64;
    int rowq = m0 + (lane & 15);
    int b = rowq / NQTOK, qi = rowq % NQTOK;
    const float* aptr = X + ((size_t)b * NTOK + sub_token(qi)) * INDIM;
    v8f acc[4] = {};
#pragma unroll 2
    for (int kk = 0; kk < INDIM; kk += 32) {
        if (kk + 32 < INDIM) __builtin_prefetch(aptr + kk + 32, 0, 1);
        v16h a = load_a_f32(aptr + kk, hi);
        v16h b0 = load_b_f32(W + (size_t)(n0 +  0 + (lane & 15)) * INDIM + kk, hi);
        v16h b1 = load_b_f32(W + (size_t)(n0 + 16 + (lane & 15)) * INDIM + kk, hi);
        v16h b2 = load_b_f32(W + (size_t)(n0 + 32 + (lane & 15)) * INDIM + kk, hi);
        v16h b3 = load_b_f32(W + (size_t)(n0 + 48 + (lane & 15)) * INDIM + kk, hi);
        acc[0] = wmma_f16(a, b0, acc[0]);
        acc[1] = wmma_f16(a, b1, acc[1]);
        acc[2] = wmma_f16(a, b2, acc[2]);
        acc[3] = wmma_f16(a, b3, acc[3]);
    }
    accum_stats(acc, lane, n0, s_sum, s_sq, g_sum, g_sq);
#pragma unroll
    for (int j = 0; j < 4; ++j) {
        int n = n0 + 16 * j + (lane & 15);
#pragma unroll
        for (int r = 0; r < 8; ++r) {
            int m = m0 + r + 8 * hi;
            out[(size_t)m * CQ + n] = (_Float16)acc[j][r];
        }
    }
}

// ---------------- flash attention per (b,h): 10 waves x 32 query rows ---------
__global__ __launch_bounds__(320)
void attn_kernel(const _Float16* __restrict__ kvb, const _Float16* __restrict__ qbf,
                 const float* __restrict__ sc_kv, const float* __restrict__ bi_kv,
                 const float* __restrict__ sc_q, const float* __restrict__ bi_q,
                 _Float16* __restrict__ obuf) {
    __shared__ _Float16 kt[32 * 32];        // [key][feat]  normalized K chunk
    __shared__ _Float16 vt[32 * 64];        // [key][d]     normalized V chunk
    __shared__ float    sS[10][16 * 32];    // per-wave raw score tile
    __shared__ _Float16 sP[10][16 * 32];    // per-wave softmaxed tile (f16, A-layout source)
    __shared__ float    rAl[10][16];        // per-wave accumulator rescale factors
    __shared__ float    rSum[10][32];       // per-wave final row sums

    int blk = blockIdx.x;
    int b = blk >> 3, h = blk & 7;
    int w = threadIdx.x >> 5, lane = threadIdx.x & 31, hi = lane >> 4;
    int q0 = w * 32;

    // ---- load + normalize Q fragments (fold softmax scale into Q) ----
    v16h aq[2];
#pragma unroll
    for (int t = 0; t < 2; ++t) {
        int rowg = b * NQTOK + q0 + 16 * t + (lane & 15);
        const _Float16* qr = qbf + (size_t)rowg * CQ + h * KD;
#pragma unroll
        for (int i = 0; i < 8; ++i) {
            int f = 16 * hi + i;
            float v = (float)qr[f] * sc_q[h * KD + f] + bi_q[h * KD + f];
            aq[t][i] = (_Float16)(v * SCALE);
            int f2 = 16 * hi + 8 + i;
            float v2 = (float)qr[f2] * sc_q[h * KD + f2] + bi_q[h * KD + f2];
            aq[t][8 + i] = (_Float16)(v2 * SCALE);
        }
    }

    float rowM = -1e30f, rowL = 0.f;        // lane <-> its own query row (m = lane)
    v8f acc[2][4] = {};

    for (int c0 = 0; c0 < NTOK; c0 += 32) {
        // ---- stage normalized K/V chunk into LDS (whole workgroup) ----
        for (int e = threadIdx.x; e < 32 * 96; e += 320) {
            int key = e / 96, cc = e % 96;
            int ch = h * 96 + cc;
            float v = (float)kvb[((size_t)(b * NTOK + c0 + key)) * CKV + ch] * sc_kv[ch] + bi_kv[ch];
            if (cc < KD) kt[key * 32 + cc] = (_Float16)v;
            else         vt[key * 64 + (cc - KD)] = (_Float16)v;
        }
        __syncthreads();

        // ---- build shared B fragments for this chunk ----
        v16h kb[2], vb[4];
#pragma unroll
        for (int jb = 0; jb < 2; ++jb) {
            const _Float16* kr = &kt[(16 * jb + (lane & 15)) * 32 + 16 * hi];
#pragma unroll
            for (int j = 0; j < 16; ++j) kb[jb][j] = kr[j];
        }
#pragma unroll
        for (int jv = 0; jv < 4; ++jv) {
            int d = 16 * jv + (lane & 15);
#pragma unroll
            for (int j = 0; j < 16; ++j) vb[jv][j] = vt[(16 * hi + j) * 64 + d];
        }

#pragma unroll
        for (int t = 0; t < 2; ++t) {
            // scores: 16 query rows x 32 keys
            v8f s0 = wmma_f16(aq[t], kb[0], v8f{});
            v8f s1 = wmma_f16(aq[t], kb[1], v8f{});
#pragma unroll
            for (int r = 0; r < 8; ++r) {
                sS[w][(r + 8 * hi) * 32 + (lane & 15)]      = s0[r];
                sS[w][(r + 8 * hi) * 32 + 16 + (lane & 15)] = s1[r];
            }
            __syncthreads();
            // online softmax: lanes whose row belongs to this 16-row tile
            if ((lane >> 4) == t) {
                int r = lane & 15;
                const float* srow = &sS[w][r * 32];
                float cmax = -1e30f;
#pragma unroll
                for (int k = 0; k < 32; ++k) cmax = fmaxf(cmax, srow[k]);
                float nm = fmaxf(rowM, cmax);
                float al = __expf(rowM - nm);
                float s = 0.f;
#pragma unroll
                for (int k = 0; k < 32; ++k) {
                    float p = __expf(srow[k] - nm);
                    s += p;
                    sP[w][r * 32 + k] = (_Float16)p;
                }
                rowL = rowL * al + s;
                rowM = nm;
                rAl[w][r] = al;
            }
            __syncthreads();
            // rescale accumulators, then P x V
            v16h ap;
            const _Float16* pr = &sP[w][(lane & 15) * 32];
#pragma unroll
            for (int i = 0; i < 8; ++i) {
                ap[i]     = pr[16 * hi + i];
                ap[8 + i] = pr[16 * hi + 8 + i];
            }
#pragma unroll
            for (int jv = 0; jv < 4; ++jv) {
#pragma unroll
                for (int r = 0; r < 8; ++r) acc[t][jv][r] *= rAl[w][r + 8 * hi];
                acc[t][jv] = wmma_f16(ap, vb[jv], acc[t][jv]);
            }
            __syncthreads();
        }
        __syncthreads();   // before restaging kt/vt
    }

    rSum[w][lane] = rowL;
    __syncthreads();
#pragma unroll
    for (int t = 0; t < 2; ++t) {
#pragma unroll
        for (int jv = 0; jv < 4; ++jv) {
#pragma unroll
            for (int r = 0; r < 8; ++r) {
                int ml = 16 * t + r + 8 * hi;
                float v = acc[t][jv][r] / rSum[w][ml];
                float u = fminf(fmaxf(v + 3.f, 0.f), 6.f);   // hard_swish
                v = v * u * (1.f / 6.f);
                int rowg = b * NQTOK + q0 + ml;
                int col = h * DV + 16 * jv + (lane & 15);
                obuf[(size_t)rowg * DH + col] = (_Float16)v;
            }
        }
    }
}

// ---------------- GEMM 3: OUTraw = O[NQ,512](f16) * W_p^T -> f32 d_out + stats
__global__ __launch_bounds__(256)
void gemm_p(const _Float16* __restrict__ A, const float* __restrict__ W,
            float* __restrict__ out, float* __restrict__ g_sum, float* __restrict__ g_sq) {
    __shared__ float s_sum[64], s_sq[64];
    int wave = threadIdx.x >> 5, lane = threadIdx.x & 31, hi = lane >> 4;
    int m0 = blockIdx.x * 128 + wave * 16;
    int n0 = blockIdx.y * 64;
    int arow = m0 + (lane & 15);
    const _Float16* aptr = A + (size_t)arow * DH;
    v8f acc[4] = {};
#pragma unroll 2
    for (int kk = 0; kk < DH; kk += 32) {
        if (kk + 32 < DH) __builtin_prefetch(aptr + kk + 32, 0, 1);
        v16h a = load_a_f16(aptr + kk, hi);
        v16h b0 = load_b_f32(W + (size_t)(n0 +  0 + (lane & 15)) * DH + kk, hi);
        v16h b1 = load_b_f32(W + (size_t)(n0 + 16 + (lane & 15)) * DH + kk, hi);
        v16h b2 = load_b_f32(W + (size_t)(n0 + 32 + (lane & 15)) * DH + kk, hi);
        v16h b3 = load_b_f32(W + (size_t)(n0 + 48 + (lane & 15)) * DH + kk, hi);
        acc[0] = wmma_f16(a, b0, acc[0]);
        acc[1] = wmma_f16(a, b1, acc[1]);
        acc[2] = wmma_f16(a, b2, acc[2]);
        acc[3] = wmma_f16(a, b3, acc[3]);
    }
    accum_stats(acc, lane, n0, s_sum, s_sq, g_sum, g_sq);
#pragma unroll
    for (int j = 0; j < 4; ++j) {
        int n = n0 + 16 * j + (lane & 15);
#pragma unroll
        for (int r = 0; r < 8; ++r) {
            int m = m0 + r + 8 * hi;
            out[(size_t)m * COUT + n] = acc[j][r];
        }
    }
}

// ---------------- host-side launch ----------------
extern "C" void kernel_launch(void* const* d_in, const int* in_sizes, int n_in,
                              void* d_out, int out_size, void* d_ws, size_t ws_size,
                              hipStream_t stream) {
    const float* x    = (const float*)d_in[0];
    const float* W_kv = (const float*)d_in[1];
    const float* g_kv = (const float*)d_in[2];
    const float* b_kv = (const float*)d_in[3];
    const float* W_q  = (const float*)d_in[4];
    const float* g_q  = (const float*)d_in[5];
    const float* b_q  = (const float*)d_in[6];
    const float* W_p  = (const float*)d_in[7];
    const float* g_p  = (const float*)d_in[8];
    const float* b_p  = (const float*)d_in[9];
    float* out = (float*)d_out;

    char* w = (char*)d_ws;
    _Float16* kvb  = (_Float16*)w;                       size_t o = (size_t)NT * CKV * 2;
    _Float16* qbf  = (_Float16*)(w + o);                 o += (size_t)NQ * CQ * 2;
    _Float16* obuf = (_Float16*)(w + o);                 o += (size_t)NQ * DH * 2;
    float* st = (float*)(w + o);
    float* sum_kv = st;            float* sq_kv = st + 768;
    float* sum_q  = st + 1536;     float* sq_q  = st + 1792;
    float* sum_p  = st + 2048;     float* sq_p  = st + 2560;
    float* sc_kv  = st + 3072;     float* bi_kv = st + 3840;
    float* sc_q   = st + 4608;     float* bi_q  = st + 4864;
    float* sc_p   = st + 5120;     float* bi_p  = st + 5632;

    zero_f32<<<12, 256, 0, stream>>>(st, 3072);
    gemm_kv<<<dim3(NT / 128, CKV / 64), 256, 0, stream>>>(x, W_kv, kvb, sum_kv, sq_kv);
    gemm_q<<<dim3(NQ / 128, CQ / 64), 256, 0, stream>>>(x, W_q, qbf, sum_q, sq_q);
    bn_finalize<<<3, 256, 0, stream>>>(sum_kv, sq_kv, g_kv, b_kv, sc_kv, bi_kv, CKV, 1.f / NT);
    bn_finalize<<<1, 256, 0, stream>>>(sum_q, sq_q, g_q, b_q, sc_q, bi_q, CQ, 1.f / NQ);
    attn_kernel<<<BATCH * HEADS, 320, 0, stream>>>(kvb, qbf, sc_kv, bi_kv, sc_q, bi_q, obuf);
    gemm_p<<<dim3(NQ / 128, COUT / 64), 256, 0, stream>>>(obuf, W_p, out, sum_p, sq_p);
    bn_finalize<<<2, 256, 0, stream>>>(sum_p, sq_p, g_p, b_p, sc_p, bi_p, COUT, 1.f / NQ);
    bn_apply<<<(NQ * COUT) / 256, 256, 0, stream>>>(out, sc_p, bi_p);
}
